// FFJORD_1786706395696
// MI455X (gfx1250) — compile-verified, matrix operand order checked by reading, not run on hardware
//
#include <hip/hip_runtime.h>
#include <hip/hip_bf16.h>
#include <math.h>

// ---------------------------------------------------------------------------
// FFJORD on MI455X (gfx1250): persistent per-block RK4 integrator.
// Each block owns 16 batch rows; 8 wave32s do WMMA f16->f32 GEMMs for the
// MLP forward + JVP tangent pass. Weights packed once to f16 B-fragments.
// N-tiles processed in pairs so one A/E LDS fragment feeds 4 WMMAs.
// ---------------------------------------------------------------------------

#define FEATD 128
#define HIDD  1024
#define NSTEPS 16

typedef __attribute__((ext_vector_type(16))) _Float16 v16h;
typedef __attribute__((ext_vector_type(8)))  _Float16 v8h;
typedef __attribute__((ext_vector_type(8)))  float    v8f;

union H16 { v16h v; v8h h[2]; };

// LDS strides (halves), padded to spread banks across the 16 A-rows.
#define XSTR 136   // layer-1 inputs (K=128)
#define HSTR 1032  // hidden activations (K=1024)

// Branch-free tanh: hardware V_TANH_F32 on gfx1250.
__device__ __forceinline__ float fast_tanh(float x) {
#if __has_builtin(__builtin_amdgcn_tanhf)
  return __builtin_amdgcn_tanhf(x);
#else
  float ax = fabsf(x);
  float e  = __expf(-2.0f * ax);
  float t  = (1.0f - e) * __frcp_rn(1.0f + e);
  return copysignf(t, x);
#endif
}

// ---------------------------------------------------------------------------
// Pack weights W[K][N] (f32, row-major) into f16 WMMA B-fragments:
// fragment (nt,kt) = 512 halves, lane-major, 16 contiguous halves per lane,
// matching the v_wmma_f32_16x16x32_f16 B operand layout.
// ---------------------------------------------------------------------------
__global__ void pack_w(const float* __restrict__ W, _Float16* __restrict__ Wp,
                       int K, int N) {
  int idx = blockIdx.x * blockDim.x + threadIdx.x;
  if (idx >= K * N) return;
  int k = idx / N, n = idx % N;
  int nt = n >> 4, nl = n & 15;
  int kt = k >> 5, kk = k & 31;
  int laneHi, h;
  if (kk < 16) { laneHi = kk >> 3; h = kk & 7; }
  else         { laneHi = (kk - 16) >> 3; h = 8 + ((kk - 16) & 7); }
  int lane = nl + (laneHi << 4);
  int nkt = K >> 5;
  size_t dst = ((size_t)(nt * nkt + kt) << 9) + (size_t)lane * 16 + h;
  Wp[dst] = (_Float16)W[(size_t)k * N + n];
}

// Load one 16x32 A-fragment from LDS (row-major f16, padded stride).
__device__ __forceinline__ v16h load_a(const _Float16* p) {
  H16 u;
  u.h[0] = *(const v8h*)(p);
  u.h[1] = *(const v8h*)(p + 16);
  return u.v;
}

// ---------------------------------------------------------------------------
// Hidden layer: H = tanh(A@W + b [+ t*w1last]) ; E = (1-H^2) * (Ae@W)
// 8 waves x 4 N-tile PAIRS cover N=1024. One A/E fragment feeds 4 WMMAs.
// ---------------------------------------------------------------------------
__device__ __forceinline__ void layer_hidden(
    const _Float16* __restrict__ Wp, int nkt,
    const float* __restrict__ bias,
    const float* __restrict__ w1last, float t,
    const _Float16* Ah, const _Float16* Ae, int astr,
    _Float16* Hout, _Float16* Eout,
    int lane, int wid) {
  int nl = lane & 15, laneHi = lane >> 4;
  const int fragOff = nkt << 9;   // halves between consecutive n-tiles
  for (int pair = 0; pair < 4; ++pair) {
    int ntIdx = wid * 8 + pair * 2;
    int n0 = ntIdx * 16 + nl;
    int n1 = n0 + 16;
    float bi0 = bias[n0], bi1 = bias[n1];
    if (w1last) { bi0 += t * w1last[n0]; bi1 += t * w1last[n1]; }
    v8f accH0, accE0, accH1, accE1;
#pragma unroll
    for (int v = 0; v < 8; ++v) {
      accH0[v] = bi0; accE0[v] = 0.f;
      accH1[v] = bi1; accE1[v] = 0.f;
    }
    const _Float16* bp = Wp + (((size_t)ntIdx * nkt) << 9) + (size_t)lane * 16;
    const _Float16* ah = Ah + nl * astr + 8 * laneHi;
    const _Float16* ae = Ae + nl * astr + 8 * laneHi;
#pragma unroll 2
    for (int kt = 0; kt < nkt; ++kt) {
      __builtin_prefetch(bp + 512, 0, 1);
      __builtin_prefetch(bp + fragOff + 512, 0, 1);
      H16 b0, b1;
      b0.h[0] = *(const v8h*)(bp);
      b0.h[1] = *(const v8h*)(bp + 8);
      b1.h[0] = *(const v8h*)(bp + fragOff);
      b1.h[1] = *(const v8h*)(bp + fragOff + 8);
      v16h a = load_a(ah);
      v16h e = load_a(ae);
      accH0 = __builtin_amdgcn_wmma_f32_16x16x32_f16(false, a, false, b0.v,
                                                     (short)0, accH0, false, false);
      accE0 = __builtin_amdgcn_wmma_f32_16x16x32_f16(false, e, false, b0.v,
                                                     (short)0, accE0, false, false);
      accH1 = __builtin_amdgcn_wmma_f32_16x16x32_f16(false, a, false, b1.v,
                                                     (short)0, accH1, false, false);
      accE1 = __builtin_amdgcn_wmma_f32_16x16x32_f16(false, e, false, b1.v,
                                                     (short)0, accE1, false, false);
      bp += 512; ah += 32; ae += 32;
    }
#pragma unroll
    for (int v = 0; v < 8; ++v) {
      int m = v + 8 * laneHi;
      float hc0 = fast_tanh(accH0[v]);
      float hc1 = fast_tanh(accH1[v]);
      float ec0 = (1.f - hc0 * hc0) * accE0[v];
      float ec1 = (1.f - hc1 * hc1) * accE1[v];
      Hout[m * HSTR + n0] = (_Float16)hc0;
      Hout[m * HSTR + n1] = (_Float16)hc1;
      Eout[m * HSTR + n0] = (_Float16)ec0;
      Eout[m * HSTR + n1] = (_Float16)ec1;
    }
  }
}

// Output layer: dx = A@W3 + b3 ; jeps = Ae@W3   (N=128 -> 1 tile per wave)
__device__ __forceinline__ void layer_out(
    const _Float16* __restrict__ Wp, int nkt, const float* __restrict__ b3,
    const _Float16* Ah, const _Float16* Ae, int astr,
    float* dxout, float* jout, int lane, int wid) {
  int nl = lane & 15, laneHi = lane >> 4;
  int n = wid * 16 + nl;
  float binit = b3[n];
  v8f accH, accE;
#pragma unroll
  for (int v = 0; v < 8; ++v) { accH[v] = binit; accE[v] = 0.f; }
  const _Float16* bp = Wp + (((size_t)wid * nkt) << 9) + (size_t)lane * 16;
  const _Float16* ah = Ah + nl * astr + 8 * laneHi;
  const _Float16* ae = Ae + nl * astr + 8 * laneHi;
#pragma unroll 4
  for (int kt = 0; kt < nkt; ++kt) {
    __builtin_prefetch(bp + 512, 0, 1);
    H16 b; b.h[0] = *(const v8h*)(bp); b.h[1] = *(const v8h*)(bp + 8);
    v16h a = load_a(ah);
    v16h e = load_a(ae);
    accH = __builtin_amdgcn_wmma_f32_16x16x32_f16(false, a, false, b.v,
                                                  (short)0, accH, false, false);
    accE = __builtin_amdgcn_wmma_f32_16x16x32_f16(false, e, false, b.v,
                                                  (short)0, accE, false, false);
    bp += 512; ah += 32; ae += 32;
  }
#pragma unroll
  for (int v = 0; v < 8; ++v) {
    int m = v + 8 * laneHi;
    dxout[m * FEATD + n] = accH[v];
    jout [m * FEATD + n] = accE[v];
  }
}

// One evaluation of the augmented dynamics: kout = vf(t, Y), trout = trace est.
__device__ __forceinline__ void eval_dyn(
    const float* Yin, float t, float* kout, float* trout,
    _Float16* xh, _Float16* epsh, _Float16* h0, _Float16* e0,
    _Float16* h1, _Float16* e1, float* jbuf, const float* epsf,
    const _Float16* W1p, const _Float16* W2p, const _Float16* W3p,
    const float* b1, const float* b2, const float* b3,
    const float* w1last, int tid) {
  // stage Y as f16 layer-1 A operand
  for (int i = tid; i < 16 * FEATD; i += 256) {
    int r = i >> 7, c = i & 127;
    xh[r * XSTR + c] = (_Float16)Yin[i];
  }
  __syncthreads();
  int lane = tid & 31, wid = tid >> 5;
  layer_hidden(W1p, 4, b1, w1last, t, xh, epsh, XSTR, h0, e0, lane, wid);
  __syncthreads();
  layer_hidden(W2p, 32, b2, nullptr, 0.f, h0, e0, HSTR, h1, e1, lane, wid);
  __syncthreads();
  layer_out(W3p, 32, b3, h1, e1, HSTR, kout, jbuf, lane, wid);
  __syncthreads();
  if (tid < 16) trout[tid] = 0.f;
  __syncthreads();
  {
    int r = tid >> 4, c0 = tid & 15;
    float p = 0.f;
#pragma unroll
    for (int j = 0; j < 8; ++j) {
      int c = c0 + 16 * j;
      p += jbuf[r * FEATD + c] * epsf[r * FEATD + c];
    }
    atomicAdd(&trout[r], p);
  }
  __syncthreads();
}

// LDS budget (bytes)
#define SMEM_F32 (2048 + 2048 + 4 * 2048 + 2048 + 2048 + 64 + 16)
#define SMEM_F16 (2 * 16 * XSTR + 4 * 16 * HSTR)
#define SMEM_BYTES (SMEM_F32 * 4 + SMEM_F16 * 2)

__global__ void __launch_bounds__(256) ffjord_rk4(
    const float* __restrict__ x, const float* __restrict__ eps,
    const float* __restrict__ W1, const float* __restrict__ b1,
    const float* __restrict__ b2, const float* __restrict__ b3,
    const _Float16* __restrict__ W1p, const _Float16* __restrict__ W2p,
    const _Float16* __restrict__ W3p,
    float* __restrict__ out, int nrows) {
  extern __shared__ char smem_raw[];
  float* yv   = (float*)smem_raw;           // 16 x 128
  float* ytmp = yv + 2048;                  // 16 x 128
  float* kbuf = ytmp + 2048;                // 4 x 16 x 128
  float* jbuf = kbuf + 4 * 2048;            // 16 x 128
  float* epsf = jbuf + 2048;                // 16 x 128
  float* trv  = epsf + 2048;                // 4 x 16
  float* ldv  = trv + 64;                   // 16
  _Float16* epsh = (_Float16*)(ldv + 16);   // 16 x XSTR
  _Float16* xh   = epsh + 16 * XSTR;        // 16 x XSTR
  _Float16* h0   = xh + 16 * XSTR;          // 16 x HSTR
  _Float16* e0   = h0 + 16 * HSTR;
  _Float16* h1   = e0 + 16 * HSTR;
  _Float16* e1   = h1 + 16 * HSTR;

  int tid = threadIdx.x;
  int r0  = blockIdx.x * 16;
  const float* w1last = W1 + (size_t)FEATD * HIDD;  // row 128 of W1 (t column)

  // load state + eps
  for (int i = tid; i < 16 * FEATD; i += 256) {
    int r = i >> 7, c = i & 127;
    yv[i] = x[(size_t)r0 * FEATD + i];
    float ev = eps[(size_t)r0 * FEATD + i];
    epsf[i] = ev;
    epsh[r * XSTR + c] = (_Float16)ev;
  }
  if (tid < 16) ldv[tid] = 0.f;
  __syncthreads();

  const float hs = 1.0f / (float)NSTEPS;
  for (int s = 0; s < NSTEPS; ++s) {
    float t0 = s * hs;
    eval_dyn(yv, t0, kbuf, trv, xh, epsh, h0, e0, h1, e1, jbuf, epsf,
             W1p, W2p, W3p, b1, b2, b3, w1last, tid);
    for (int i = tid; i < 2048; i += 256) ytmp[i] = yv[i] + 0.5f * hs * kbuf[i];
    __syncthreads();
    eval_dyn(ytmp, t0 + 0.5f * hs, kbuf + 2048, trv + 16, xh, epsh, h0, e0, h1,
             e1, jbuf, epsf, W1p, W2p, W3p, b1, b2, b3, w1last, tid);
    for (int i = tid; i < 2048; i += 256) ytmp[i] = yv[i] + 0.5f * hs * kbuf[2048 + i];
    __syncthreads();
    eval_dyn(ytmp, t0 + 0.5f * hs, kbuf + 4096, trv + 32, xh, epsh, h0, e0, h1,
             e1, jbuf, epsf, W1p, W2p, W3p, b1, b2, b3, w1last, tid);
    for (int i = tid; i < 2048; i += 256) ytmp[i] = yv[i] + hs * kbuf[4096 + i];
    __syncthreads();
    eval_dyn(ytmp, t0 + hs, kbuf + 6144, trv + 48, xh, epsh, h0, e0, h1, e1,
             jbuf, epsf, W1p, W2p, W3p, b1, b2, b3, w1last, tid);
    for (int i = tid; i < 2048; i += 256)
      yv[i] += (hs / 6.f) * (kbuf[i] + 2.f * kbuf[2048 + i] +
                             2.f * kbuf[4096 + i] + kbuf[6144 + i]);
    if (tid < 16)
      ldv[tid] -= (hs / 6.f) * (trv[tid] + 2.f * trv[16 + tid] +
                                2.f * trv[32 + tid] + trv[48 + tid]);
    __syncthreads();
  }

  // write z then log_det (concatenated flat, return order)
  for (int i = tid; i < 2048; i += 256) out[(size_t)r0 * FEATD + i] = yv[i];
  if (tid < 16) out[(size_t)nrows * FEATD + r0 + tid] = ldv[tid];
}

extern "C" void kernel_launch(void* const* d_in, const int* in_sizes, int n_in,
                              void* d_out, int out_size, void* d_ws, size_t ws_size,
                              hipStream_t stream) {
  const float* x   = (const float*)d_in[0];
  const float* eps = (const float*)d_in[1];
  const float* W1  = (const float*)d_in[2];
  const float* b1  = (const float*)d_in[3];
  const float* W2  = (const float*)d_in[4];
  const float* b2  = (const float*)d_in[5];
  const float* W3  = (const float*)d_in[6];
  const float* b3  = (const float*)d_in[7];

  _Float16* W1p = (_Float16*)d_ws;                 // 128*1024 halves
  _Float16* W2p = W1p + 128 * 1024;                // 1024*1024 halves
  _Float16* W3p = W2p + 1024 * 1024;               // 1024*128 halves

  pack_w<<<(128 * 1024 + 255) / 256, 256, 0, stream>>>(W1, W1p, 128, 1024);
  pack_w<<<(1024 * 1024 + 255) / 256, 256, 0, stream>>>(W2, W2p, 1024, 1024);
  pack_w<<<(1024 * 128 + 255) / 256, 256, 0, stream>>>(W3, W3p, 1024, 128);

  int nrows = in_sizes[0] / FEATD;   // 16384
  int blocks = nrows / 16;           // 1024
  ffjord_rk4<<<blocks, 256, SMEM_BYTES, stream>>>(x, eps, W1, b1, b2, b3, W1p,
                                                  W2p, W3p, (float*)d_out, nrows);
}
